// TorchMD_GN_20779051778081
// MI455X (gfx1250) — matrix operand order repaired and checked by reading, not compiled, not run on hardware
//
#include <hip/hip_runtime.h>
#include <hip/hip_bf16.h>

#define E_EDGES 640000
#define NN 20000
#define HID 128
#define RBF 50
#define RBFP 64
#define NLAYERS 6
#define PI_F 3.14159265358979323846f

typedef __attribute__((ext_vector_type(16))) __bf16 v16bf;
typedef __attribute__((ext_vector_type(8)))  float  v8f;

// ---------- gfx1250 async global->LDS staging (ASYNCcnt path) ----------
#if defined(__has_builtin)
#if __has_builtin(__builtin_amdgcn_global_load_async_to_lds_b128)
#define HAVE_ASYNC_COPY 1
#endif
#endif

#ifdef HAVE_ASYNC_COPY
// Signature (from compiler diagnostic): param0 = v4i __device__ (as1) *,
// i.e. (global v4i* src, lds v4i* dst, imm offset, imm cpol).
typedef int v4i_ __attribute__((vector_size(16)));
typedef __attribute__((address_space(1))) v4i_* gv4i_p;
typedef __attribute__((address_space(3))) v4i_* lv4i_p;
#endif

__device__ __forceinline__ void async_copy_b128(const void* g, void* l) {
#ifdef HAVE_ASYNC_COPY
  __builtin_amdgcn_global_load_async_to_lds_b128(
      (gv4i_p)(uintptr_t)g, (lv4i_p)(unsigned)(uintptr_t)l, 0, 0);
#else
  *(uint4*)l = *(const uint4*)g;
#endif
}

__device__ __forceinline__ void async_copy_wait() {
#ifdef HAVE_ASYNC_COPY
  asm volatile("s_wait_asynccnt 0" ::: "memory");
#endif
}

// ---- WMMA fragment loaders (CDNA5 wave32 layouts, 16x16x32 bf16) ----
// A (16xK row-major): lane L holds row L%16; K-chunks [half*8, half*8+8) and
// [16+half*8, 16+half*8+8) within each 32-wide K step (half = L/16).
__device__ __forceinline__ v16bf load_a_frag(const __bf16* A, int stride, int kbase) {
  int lane = threadIdx.x & 31;
  int row  = lane & 15;
  int half = lane >> 4;
  const __bf16* p = A + (size_t)row * stride + kbase;
  v16bf a;
#pragma unroll
  for (int i = 0; i < 8; ++i) a[i] = p[half * 8 + i];
#pragma unroll
  for (int i = 0; i < 8; ++i) a[8 + i] = p[16 + half * 8 + i];
  return a;
}

// B (Kx16): lane L holds column L%16, K-chunks [half*8, half*8+8) and
// [16+half*8, 16+half*8+8) of the 32-K step. Weights are stored transposed: Wt[col][Kp].
__device__ __forceinline__ v16bf load_b_frag(const __bf16* Wt, int Kp, int ntile, int kbase) {
  int lane = threadIdx.x & 31;
  int col  = lane & 15;
  int half = lane >> 4;
  const __bf16* p = Wt + (size_t)(ntile * 16 + col) * Kp + kbase;
  v16bf b;
#pragma unroll
  for (int i = 0; i < 8; ++i) b[i] = p[half * 8 + i];
#pragma unroll
  for (int i = 0; i < 8; ++i) b[8 + i] = p[16 + half * 8 + i];
  return b;
}

#define WMMA_BF16(a, b, c) \
  __builtin_amdgcn_wmma_f32_16x16x32_bf16(false, (a), false, (b), (short)0, (c), false, false)

__device__ __forceinline__ float silu_f(float t) { return t / (1.0f + expf(-t)); }

// ---- weight transpose + fp32->bf16 convert, K padded to Kp with zeros ----
__global__ void k_transpose_convert(const float* __restrict__ in, __bf16* __restrict__ out,
                                    int K, int Kp, int N, int total) {
  int idx = blockIdx.x * blockDim.x + threadIdx.x;
  if (idx >= total) return;
  int k = idx % Kp;
  int rest = idx / Kp;
  int n = rest % N;
  int mat = rest / N;
  float v = (k < K) ? in[(size_t)mat * K * N + (size_t)k * N + n] : 0.0f;
  out[idx] = (__bf16)v;
}

// ---- per-edge distance, cutoff, expnorm RBF (bf16, padded to 64) ----
__global__ void k_edge_rbf(const int* __restrict__ ei, const float* __restrict__ pos,
                           const float* __restrict__ mu, const float* __restrict__ beta,
                           __bf16* __restrict__ ea, float* __restrict__ C,
                           float* __restrict__ Cm) {
  int e = blockIdx.x * blockDim.x + threadIdx.x;
  if (e >= E_EDGES) return;
  int s = ei[e], d = ei[E_EDGES + e];
  float dx = pos[s * 3 + 0] - pos[d * 3 + 0];
  float dy = pos[s * 3 + 1] - pos[d * 3 + 1];
  float dz = pos[s * 3 + 2] - pos[d * 3 + 2];
  float dist = sqrtf(dx * dx + dy * dy + dz * dz);
  float c = 0.5f * (cosf(dist * (PI_F / 5.0f)) + 1.0f);
  c = (dist < 5.0f) ? c : 0.0f;
  C[e] = c;
  Cm[e] = (s != d) ? c : 0.0f;
  float ex = expf(-dist);  // alpha = 5/(5-0) = 1, cutoff_lower = 0
  __bf16* row = ea + (size_t)e * RBFP;
  for (int r = 0; r < RBF; ++r) {
    float t = ex - mu[r];
    row[r] = (__bf16)(c * expf(-beta[r] * t * t));
  }
#pragma unroll
  for (int r = RBF; r < RBFP; ++r) row[r] = (__bf16)0.0f;
}

// ---- x = emb_table[z]; ag = 0 ----
__global__ void k_embed(const int* __restrict__ z, const float* __restrict__ tab,
                        float* __restrict__ x, float* __restrict__ ag) {
  int idx = blockIdx.x * blockDim.x + threadIdx.x;
  if (idx >= NN * HID) return;
  int n = idx >> 7;
  int c = idx & 127;
  x[idx] = tab[(size_t)z[n] * HID + c];
  ag[idx] = 0.0f;
}

// ---- NeighborEmbedding: Wn = (ea @ nbrW + b) * Cm ; ag[dst] += xn[src]*Wn ----
__global__ void __launch_bounds__(256) k_nbr_message(
    const __bf16* __restrict__ ea, const __bf16* __restrict__ Wt,
    const float* __restrict__ bias, const float* __restrict__ Cm,
    const int* __restrict__ ei, const int* __restrict__ z,
    const float* __restrict__ nbr_tab, float* __restrict__ ag) {
  __shared__ __bf16 sW[HID * RBFP];  // 16 KB staged weights
  int tid = threadIdx.x;
  {
    const char* gw = (const char*)Wt;
    char* lw = (char*)sW;
#pragma unroll
    for (int i = 0; i < (HID * RBFP * 2) / 16 / 256; ++i)  // 4 iters
      async_copy_b128(gw + (tid + i * 256) * 16, lw + (tid + i * 256) * 16);
    async_copy_wait();
  }
  __syncthreads();

  int wid = tid >> 5;
  int tile = blockIdx.x * 8 + wid;  // 16 edges per wave
  int lane = tid & 31;
  int col = lane & 15, half = lane >> 4;
  const __bf16* A = ea + (size_t)tile * 16 * RBFP;
  v16bf a0 = load_a_frag(A, RBFP, 0);
  v16bf a1 = load_a_frag(A, RBFP, 32);
  int dsts[8], xoff[8];
  float cms[8];
#pragma unroll
  for (int v = 0; v < 8; ++v) {
    int e = tile * 16 + v + 8 * half;  // D row m = v + 8*half
    int s = ei[e];
    dsts[v] = ei[E_EDGES + e];
    cms[v] = Cm[e];
    xoff[v] = z[s] * HID;
  }
#pragma unroll
  for (int nt = 0; nt < 8; ++nt) {
    v8f acc = {};
    v16bf b0 = load_b_frag(sW, RBFP, nt, 0);
    acc = WMMA_BF16(a0, b0, acc);
    v16bf b1 = load_b_frag(sW, RBFP, nt, 32);
    acc = WMMA_BF16(a1, b1, acc);
    int c = nt * 16 + col;
    float bb = bias[c];
#pragma unroll
    for (int v = 0; v < 8; ++v) {
      float val = (acc[v] + bb) * cms[v];
      float xn = nbr_tab[xoff[v] + c];
      unsafeAtomicAdd(&ag[(size_t)dsts[v] * HID + c], val * xn);
    }
  }
}

// ---- xa = bf16(concat(x, ag)) ----
__global__ void k_xa_build(const float* __restrict__ x, const float* __restrict__ ag,
                           __bf16* __restrict__ xa) {
  int idx = blockIdx.x * blockDim.x + threadIdx.x;
  if (idx >= NN * 256) return;
  int n = idx >> 8;
  int k = idx & 255;
  float v = (k < HID) ? x[(size_t)n * HID + k] : ag[(size_t)n * HID + (k - HID)];
  xa[idx] = (__bf16)v;
}

// ---- x = xa @ combine_W + b  (K = 256) ----
__global__ void __launch_bounds__(64) k_combine(
    const __bf16* __restrict__ xa, const __bf16* __restrict__ Wt,
    const float* __restrict__ bias, float* __restrict__ x) {
  int wid = threadIdx.x >> 5;
  int tile = blockIdx.x * 2 + wid;
  int lane = threadIdx.x & 31;
  int col = lane & 15, half = lane >> 4;
  const __bf16* A = xa + (size_t)tile * 16 * 256;
  v16bf a[8];
#pragma unroll
  for (int ks = 0; ks < 8; ++ks) a[ks] = load_a_frag(A, 256, ks * 32);
#pragma unroll
  for (int nt = 0; nt < 8; ++nt) {
    v8f acc = {};
#pragma unroll
    for (int ks = 0; ks < 8; ++ks) {
      v16bf b = load_b_frag(Wt, 256, nt, ks * 32);
      acc = WMMA_BF16(a[ks], b, acc);
    }
    int c = nt * 16 + col;
    float bb = bias[c];
#pragma unroll
    for (int v = 0; v < 8; ++v)
      x[(size_t)(tile * 16 + v + 8 * half) * HID + c] = acc[v] + bb;
  }
}

// ---- xb = bf16(x); ag = 0 ----
__global__ void k_state_bf16_zero(const float* __restrict__ x, __bf16* __restrict__ xb,
                                  float* __restrict__ ag) {
  int idx = blockIdx.x * blockDim.x + threadIdx.x;
  if (idx >= NN * HID) return;
  xb[idx] = (__bf16)x[idx];
  ag[idx] = 0.0f;
}

// ---- h = x @ lin1_W (no bias) ----
__global__ void __launch_bounds__(64) k_lin1(const __bf16* __restrict__ xb,
                                             const __bf16* __restrict__ Wt,
                                             float* __restrict__ h) {
  int wid = threadIdx.x >> 5;
  int tile = blockIdx.x * 2 + wid;
  int lane = threadIdx.x & 31;
  int col = lane & 15, half = lane >> 4;
  const __bf16* A = xb + (size_t)tile * 16 * HID;
  v16bf a[4];
#pragma unroll
  for (int ks = 0; ks < 4; ++ks) a[ks] = load_a_frag(A, HID, ks * 32);
#pragma unroll
  for (int nt = 0; nt < 8; ++nt) {
    v8f acc = {};
#pragma unroll
    for (int ks = 0; ks < 4; ++ks) {
      v16bf b = load_b_frag(Wt, HID, nt, ks * 32);
      acc = WMMA_BF16(a[ks], b, acc);
    }
    int c = nt * 16 + col;
#pragma unroll
    for (int v = 0; v < 8; ++v)
      h[(size_t)(tile * 16 + v + 8 * half) * HID + c] = acc[v];
  }
}

// ---- fused CFConv: Wf = (silu(ea@W1+b1)@W2+b2)*C ; ag[dst] += h[src]*Wf ----
__global__ void __launch_bounds__(256) k_cfconv_message(
    const __bf16* __restrict__ ea, const __bf16* __restrict__ W1t,
    const float* __restrict__ b1, const __bf16* __restrict__ W2t,
    const float* __restrict__ b2, const float* __restrict__ C,
    const int* __restrict__ ei, const float* __restrict__ h,
    float* __restrict__ ag) {
  __shared__ __bf16 g1[8][16 * HID];   // silu intermediate, bf16, per wave (32 KB)
  __shared__ __bf16 sW1[HID * RBFP];   // staged W1^T (16 KB)
  __shared__ __bf16 sW2[HID * HID];    // staged W2^T (32 KB)
  int tid = threadIdx.x;
  {
    const char* gw1 = (const char*)W1t;
    const char* gw2 = (const char*)W2t;
    char* lw1 = (char*)sW1;
    char* lw2 = (char*)sW2;
#pragma unroll
    for (int i = 0; i < (HID * RBFP * 2) / 16 / 256; ++i)  // 4 iters
      async_copy_b128(gw1 + (tid + i * 256) * 16, lw1 + (tid + i * 256) * 16);
#pragma unroll
    for (int i = 0; i < (HID * HID * 2) / 16 / 256; ++i)   // 8 iters
      async_copy_b128(gw2 + (tid + i * 256) * 16, lw2 + (tid + i * 256) * 16);
    async_copy_wait();
  }
  __syncthreads();

  int wid = tid >> 5;
  int tile = blockIdx.x * 8 + wid;
  int lane = tid & 31;
  int col = lane & 15, half = lane >> 4;
  const __bf16* A = ea + (size_t)tile * 16 * RBFP;
  v16bf a0 = load_a_frag(A, RBFP, 0);
  v16bf a1 = load_a_frag(A, RBFP, 32);
  int srcs[8], dsts[8];
  float cc[8];
#pragma unroll
  for (int v = 0; v < 8; ++v) {
    int e = tile * 16 + v + 8 * half;
    srcs[v] = ei[e];
    dsts[v] = ei[E_EDGES + e];
    cc[v] = C[e];
  }
  // GEMM 1: [16,64] x [64,128] -> silu -> LDS (bf16)
#pragma unroll
  for (int nt = 0; nt < 8; ++nt) {
    v8f acc = {};
    v16bf b = load_b_frag(sW1, RBFP, nt, 0);
    acc = WMMA_BF16(a0, b, acc);
    b = load_b_frag(sW1, RBFP, nt, 32);
    acc = WMMA_BF16(a1, b, acc);
    int c = nt * 16 + col;
    float bb = b1[c];
#pragma unroll
    for (int v = 0; v < 8; ++v)
      g1[wid][(v + 8 * half) * HID + c] = (__bf16)silu_f(acc[v] + bb);
  }
  __syncthreads();
  // GEMM 2: [16,128] x [128,128], then gather h[src] and scatter to ag[dst]
  v16bf a[4];
#pragma unroll
  for (int ks = 0; ks < 4; ++ks) a[ks] = load_a_frag(&g1[wid][0], HID, ks * 32);
#pragma unroll
  for (int nt = 0; nt < 8; ++nt) {
    v8f acc = {};
#pragma unroll
    for (int ks = 0; ks < 4; ++ks) {
      v16bf b = load_b_frag(sW2, HID, nt, ks * 32);
      acc = WMMA_BF16(a[ks], b, acc);
    }
    int c = nt * 16 + col;
    float bb = b2[c];
#pragma unroll
    for (int v = 0; v < 8; ++v) {
      float wf = (acc[v] + bb) * cc[v];
      float hv = h[(size_t)srcs[v] * HID + c];
      unsafeAtomicAdd(&ag[(size_t)dsts[v] * HID + c], wf * hv);
    }
  }
}

// ---- node update: x += silu(ag@l2W + l2b) @ lW + lb ----
__global__ void __launch_bounds__(64) k_node_update(
    const float* __restrict__ ag, const __bf16* __restrict__ l2t,
    const float* __restrict__ l2b, const __bf16* __restrict__ lWt,
    const float* __restrict__ lb, float* __restrict__ x) {
  __shared__ __bf16 sA[2][16 * HID];
  __shared__ __bf16 sU[2][16 * HID];
  int wid = threadIdx.x >> 5;
  int tile = blockIdx.x * 2 + wid;
  int lane = threadIdx.x & 31;
  int col = lane & 15, half = lane >> 4;
  for (int i = lane; i < 16 * HID; i += 32)
    sA[wid][i] = (__bf16)ag[(size_t)tile * 16 * HID + i];
  __syncthreads();
  v16bf a[4];
#pragma unroll
  for (int ks = 0; ks < 4; ++ks) a[ks] = load_a_frag(&sA[wid][0], HID, ks * 32);
#pragma unroll
  for (int nt = 0; nt < 8; ++nt) {
    v8f acc = {};
#pragma unroll
    for (int ks = 0; ks < 4; ++ks) {
      v16bf b = load_b_frag(l2t, HID, nt, ks * 32);
      acc = WMMA_BF16(a[ks], b, acc);
    }
    int c = nt * 16 + col;
    float bb = l2b[c];
#pragma unroll
    for (int v = 0; v < 8; ++v)
      sU[wid][(v + 8 * half) * HID + c] = (__bf16)silu_f(acc[v] + bb);
  }
  __syncthreads();
#pragma unroll
  for (int ks = 0; ks < 4; ++ks) a[ks] = load_a_frag(&sU[wid][0], HID, ks * 32);
#pragma unroll
  for (int nt = 0; nt < 8; ++nt) {
    v8f acc = {};
#pragma unroll
    for (int ks = 0; ks < 4; ++ks) {
      v16bf b = load_b_frag(lWt, HID, nt, ks * 32);
      acc = WMMA_BF16(a[ks], b, acc);
    }
    int c = nt * 16 + col;
    float bb = lb[c];
#pragma unroll
    for (int v = 0; v < 8; ++v) {
      size_t o = (size_t)(tile * 16 + v + 8 * half) * HID + c;
      x[o] = x[o] + acc[v] + bb;
    }
  }
}

extern "C" void kernel_launch(void* const* d_in, const int* in_sizes, int n_in,
                              void* d_out, int out_size, void* d_ws, size_t ws_size,
                              hipStream_t stream) {
  (void)in_sizes; (void)n_in; (void)out_size; (void)ws_size;
  const int*   z       = (const int*)d_in[0];
  const float* pos     = (const float*)d_in[1];
  const int*   ei      = (const int*)d_in[2];
  const float* emb     = (const float*)d_in[4];
  const float* nbr_tab = (const float*)d_in[5];
  const float* nbr_W   = (const float*)d_in[6];
  const float* nbr_b   = (const float*)d_in[7];
  const float* comb_W  = (const float*)d_in[8];
  const float* comb_b  = (const float*)d_in[9];
  const float* mu      = (const float*)d_in[10];
  const float* beta    = (const float*)d_in[11];
  const float* W1      = (const float*)d_in[12];
  const float* b1      = (const float*)d_in[13];
  const float* W2      = (const float*)d_in[14];
  const float* b2      = (const float*)d_in[15];
  const float* l1W     = (const float*)d_in[16];
  const float* l2W     = (const float*)d_in[17];
  const float* l2b     = (const float*)d_in[18];
  const float* lW      = (const float*)d_in[19];
  const float* lb      = (const float*)d_in[20];
  float* x = (float*)d_out;

  char* ws = (char*)d_ws;
  size_t off = 0;
  auto walloc = [&](size_t bytes) -> void* {
    void* p = ws + off;
    off = (off + bytes + 255) & ~(size_t)255;
    return p;
  };
  __bf16* ea     = (__bf16*)walloc((size_t)E_EDGES * RBFP * 2);
  float*  C      = (float*) walloc((size_t)E_EDGES * 4);
  float*  Cm     = (float*) walloc((size_t)E_EDGES * 4);
  float*  ag     = (float*) walloc((size_t)NN * HID * 4);
  float*  h      = (float*) walloc((size_t)NN * HID * 4);
  __bf16* xb     = (__bf16*)walloc((size_t)NN * HID * 2);
  __bf16* xa     = (__bf16*)walloc((size_t)NN * 256 * 2);
  __bf16* nbrWt  = (__bf16*)walloc((size_t)128 * 64 * 2);
  __bf16* combWt = (__bf16*)walloc((size_t)128 * 256 * 2);
  __bf16* W1t    = (__bf16*)walloc((size_t)6 * 128 * 64 * 2);
  __bf16* W2t    = (__bf16*)walloc((size_t)6 * 128 * 128 * 2);
  __bf16* l1t    = (__bf16*)walloc((size_t)6 * 128 * 128 * 2);
  __bf16* l2t    = (__bf16*)walloc((size_t)6 * 128 * 128 * 2);
  __bf16* lWt    = (__bf16*)walloc((size_t)6 * 128 * 128 * 2);

  auto cdiv = [](int a, int b) { return (a + b - 1) / b; };

  // weight conversion (transpose to Wt[col][Kp], bf16, zero-padded K)
  k_transpose_convert<<<cdiv(128 * 64, 256), 256, 0, stream>>>(nbr_W, nbrWt, 50, 64, 128, 128 * 64);
  k_transpose_convert<<<cdiv(128 * 256, 256), 256, 0, stream>>>(comb_W, combWt, 256, 256, 128, 128 * 256);
  k_transpose_convert<<<cdiv(6 * 128 * 64, 256), 256, 0, stream>>>(W1, W1t, 50, 64, 128, 6 * 128 * 64);
  k_transpose_convert<<<cdiv(6 * 128 * 128, 256), 256, 0, stream>>>(W2, W2t, 128, 128, 128, 6 * 128 * 128);
  k_transpose_convert<<<cdiv(6 * 128 * 128, 256), 256, 0, stream>>>(l1W, l1t, 128, 128, 128, 6 * 128 * 128);
  k_transpose_convert<<<cdiv(6 * 128 * 128, 256), 256, 0, stream>>>(l2W, l2t, 128, 128, 128, 6 * 128 * 128);
  k_transpose_convert<<<cdiv(6 * 128 * 128, 256), 256, 0, stream>>>(lW, lWt, 128, 128, 128, 6 * 128 * 128);

  // edge features + embeddings + neighbor embedding
  k_edge_rbf<<<E_EDGES / 256, 256, 0, stream>>>(ei, pos, mu, beta, ea, C, Cm);
  k_embed<<<(NN * HID) / 256, 256, 0, stream>>>(z, emb, x, ag);
  k_nbr_message<<<E_EDGES / 16 / 8, 256, 0, stream>>>(ea, nbrWt, nbr_b, Cm, ei, z, nbr_tab, ag);
  k_xa_build<<<(NN * 256) / 256, 256, 0, stream>>>(x, ag, xa);
  k_combine<<<NN / 16 / 2, 64, 0, stream>>>(xa, combWt, comb_b, x);

  // 6 interaction layers
  for (int l = 0; l < NLAYERS; ++l) {
    k_state_bf16_zero<<<(NN * HID) / 256, 256, 0, stream>>>(x, xb, ag);
    k_lin1<<<NN / 16 / 2, 64, 0, stream>>>(xb, l1t + (size_t)l * 128 * 128, h);
    k_cfconv_message<<<E_EDGES / 16 / 8, 256, 0, stream>>>(
        ea, W1t + (size_t)l * 128 * 64, b1 + (size_t)l * 128,
        W2t + (size_t)l * 128 * 128, b2 + (size_t)l * 128, C, ei, h, ag);
    k_node_update<<<NN / 16 / 2, 64, 0, stream>>>(
        ag, l2t + (size_t)l * 128 * 128, l2b + (size_t)l * 128,
        lWt + (size_t)l * 128 * 128, lb + (size_t)l * 128, x);
  }
}
